// Policy_59468117181068
// MI455X (gfx1250) — matrix operand order repaired and checked by reading, not compiled, not run on hardware
//
#include <hip/hip_runtime.h>
#include <hip/hip_bf16.h>

#define N_NODES 1024
#define KDEG    32
#define FDIM    8192
#define EDIM    64
#define H1DIM   128
#define H2DIM   64
#define TSTEPS  3

typedef __bf16 v16bf __attribute__((ext_vector_type(16)));
typedef float  v8f   __attribute__((ext_vector_type(8)));
typedef unsigned int v4u __attribute__((ext_vector_type(4)));
typedef int          v8i __attribute__((ext_vector_type(8)));
typedef int          v4i __attribute__((ext_vector_type(4)));

// native f32 -> bf16 conversion (backend emits packed cvt on gfx1250)
__device__ __forceinline__ __bf16 f2bf(float f) { return (__bf16)f; }

__device__ __forceinline__ v8f bf16_wmma(v16bf a, v16bf b, v8f c) {
    // D = A(16x32 bf16) * B(32x16 bf16) + C(16x16 f32)
    return __builtin_amdgcn_wmma_f32_16x16x32_bf16(
        /*neg_a=*/false, a, /*neg_b=*/false, b,
        /*c_mod=*/(short)0, c, /*reuse_a=*/false, /*reuse_b=*/false);
}

// ---------------------------------------------------------------------------
// Tensor Data Mover: 2D tile (f32) global -> LDS, per CDNA5 D# layout.
//   group0: [1:0]=count=1, [63:32]=lds_addr, [120:64]=global_addr, [127:126]=type=2
//   group1: data_size=2(4B), tensor_dim0/1, tile_dim0/1, tensor_dim0_stride
// 6-arg builtin form (clang-23 / therock-10.0 toolchain).
// ---------------------------------------------------------------------------
__device__ __forceinline__ void tdm_load_2d_f32(unsigned lds_off, const void* gaddr,
                                                unsigned tile_d0, unsigned tile_d1,
                                                unsigned tensor_d0, unsigned tensor_d1,
                                                unsigned stride_d0) {
    unsigned long long ga = (unsigned long long)gaddr;
    v4u g0;
    g0[0] = 1u;                                                   // count=1
    g0[1] = lds_off;                                              // lds_addr (bytes)
    g0[2] = (unsigned)(ga & 0xFFFFFFFFu);                         // global_addr[31:0]
    g0[3] = (unsigned)((ga >> 32) & 0x01FFFFFFu) | 0x80000000u;   // ga[56:32] | type=2
    v8i g1;
    g1[0] = (int)(2u << 16);                                      // data_size = 4 bytes
    g1[1] = (int)((tensor_d0 & 0xFFFFu) << 16);                   // tensor_dim0[15:0]
    g1[2] = (int)(((tensor_d0 >> 16) & 0xFFFFu) |
                  ((tensor_d1 & 0xFFFFu) << 16));                 // td0 hi | td1 lo
    g1[3] = (int)(((tensor_d1 >> 16) & 0xFFFFu) |
                  ((tile_d0  & 0xFFFFu) << 16));                  // td1 hi | tile_dim0
    g1[4] = (int)(tile_d1 & 0xFFFFu);                             // tile_dim1 (tile_dim2=0)
    g1[5] = (int)stride_d0;                                       // tensor_dim0_stride lo
    g1[6] = 0;
    g1[7] = 0;
    v4i z4 = {};
    v8i z8 = {};
    __builtin_amdgcn_tensor_load_to_lds(g0, g1, z4, z4, z8, 0);
}

// A fragment (16x32, bf16) from a row-major bf16 LDS matrix, leading dim ld.
__device__ __forceinline__ v16bf ldsA(const __bf16* mat, int ld, int m0, int k0, int lane) {
    int m  = m0 + (lane & 15);
    int kb = k0 + ((lane >= 16) ? 16 : 0);
    v16bf a;
    #pragma unroll
    for (int j = 0; j < 16; ++j) {
        int k = kb + j;
        a[j] = mat[m * ld + k];
    }
    return a;
}

// B fragment (32x16, bf16) from a row-major bf16 LDS matrix [K x N], leading dim ld.
__device__ __forceinline__ v16bf ldsB(const __bf16* mat, int ld, int k0, int n0, int lane) {
    int kb = k0 + ((lane >= 16) ? 16 : 0);
    int n  = n0 + (lane & 15);
    v16bf b;
    #pragma unroll
    for (int j = 0; j < 16; ++j) b[j] = mat[(kb + j) * ld + n];
    return b;
}

// ---------------------------------------------------------------------------
// Kernel 1: feat0 = relu(X[N,F] @ W_emb[F,E] + b_emb)
// One wave32 per 16x16 tile; A tiles streamed by the TDM (double-buffered LDS).
// ---------------------------------------------------------------------------
__global__ __launch_bounds__(32) void embed_gemm(const float* __restrict__ X,
                                                 const float* __restrict__ W,
                                                 const float* __restrict__ bias,
                                                 float* __restrict__ out) {
    __shared__ float sA[2][16 * 32];        // only shared object -> LDS offset 0
    const unsigned ldsBase = (unsigned)(unsigned long long)(void*)&sA[0][0];

    const int bid = blockIdx.x;
    const int m0  = (bid >> 2) * 16;        // 64 row tiles
    const int n0  = (bid & 3) * 16;         // 4 col tiles
    const int l   = threadIdx.x;

    // prologue: TDM fetch of A tile 0 (rows m0..m0+15, cols 0..31)
    tdm_load_2d_f32(ldsBase, &X[(size_t)m0 * FDIM], 32, 16, FDIM, N_NODES, FDIM);

    v8f acc = {};
    const int n = n0 + (l & 15);
    for (int ks = 0; ks < FDIM / 32; ++ks) {
        // issue next tile into the other buffer while computing this one
        if (ks + 1 < FDIM / 32) {
            tdm_load_2d_f32(ldsBase + ((unsigned)(ks + 1) & 1u) * 2048u,
                            &X[(size_t)m0 * FDIM + (size_t)(ks + 1) * 32],
                            32, 16, FDIM, N_NODES, FDIM);
            __builtin_amdgcn_s_wait_tensorcnt(1);   // current tile landed
        } else {
            __builtin_amdgcn_s_wait_tensorcnt(0);
        }
        const float* At = sA[ks & 1];
        v16bf a, b;
        const int kbA = (l >= 16) ? 16 : 0;
        #pragma unroll
        for (int j = 0; j < 16; ++j) {
            int k = kbA + j;
            a[j] = f2bf(At[(l & 15) * 32 + k]);
        }
        const int kbB = ks * 32 + ((l >= 16) ? 16 : 0);
        __builtin_prefetch(&W[(size_t)(kbB + 64) * EDIM + n], 0, 1);
        #pragma unroll
        for (int j = 0; j < 16; ++j) b[j] = f2bf(W[(size_t)(kbB + j) * EDIM + n]);
        acc = bf16_wmma(a, b, acc);
    }
    const float bv = bias[n];
    const int mb = m0 + ((l >= 16) ? 8 : 0);
    #pragma unroll
    for (int r = 0; r < 8; ++r) {
        float v = acc[r] + bv;
        out[(size_t)(mb + r) * EDIM + n] = v > 0.f ? v : 0.f;
    }
}

// ---------------------------------------------------------------------------
// Kernel 2: sequential node scan. Single workgroup, 4 waves (128 threads).
// ---------------------------------------------------------------------------
__global__ __launch_bounds__(128) void policy_scan(
        const float* __restrict__ X,        // feature_origin [N,F]
        const int*   __restrict__ nbr,      // neighbors [N,K]
        const float* __restrict__ Wemb, const float* __restrict__ bemb,
        const float* __restrict__ W1,   const float* __restrict__ b1,
        const float* __restrict__ W2,   const float* __restrict__ b2,
        const float* __restrict__ Wlk,  const float* __restrict__ blk,
        const float* __restrict__ Wact, const float* __restrict__ bact,
        float* __restrict__ feat,           // [N,E] working feature (= output)
        float* __restrict__ sumhv)          // [F] scratch (d_ws)
{
    __shared__ __bf16 sW1[H1DIM * H1DIM];   // 32 KB  [k][n]
    __shared__ __bf16 sS [48 * H1DIM];      // 12 KB  S, reused for relu(H1)
    __shared__ float  sH2[48 * H2DIM];      // 12 KB
    __shared__ __bf16 sHv[FDIM];            // 16 KB  hv_mean (bf16)
    __shared__ __bf16 sUo[FDIM];            // 16 KB  u_orig  (bf16)
    __shared__ float  sLk[KDEG + 1];
    __shared__ int    sValid[KDEG];
    __shared__ int    sU, sUpd, sDone;
    __shared__ float  sTake, sInvDen, sCnt;

    const int tid = threadIdx.x;
    const int w   = tid >> 5;
    const int l   = tid & 31;

    for (int p = tid; p < H1DIM * H1DIM; p += 128) sW1[p] = f2bf(W1[p]);
    __syncthreads();

    for (int i = 0; i < N_NODES; ++i) {
        for (int f = tid; f < FDIM; f += 128) sumhv[f] = 0.f;
        if (tid < KDEG) sValid[tid] = 1;
        if (tid == 0) { sCnt = 0.f; sDone = 0; }
        __syncthreads();
        const int* idx = nbr + i * KDEG;

        for (int t = 0; t < TSTEPS; ++t) {
            // ---- build S[48,128]: rows 0..31 = [feat_i | cand_r], row 32 = [feat_i | 0] ----
            for (int r = 0; r < KDEG + 1; ++r) {
                float v;
                if (tid < EDIM) {
                    v = feat[(size_t)i * EDIM + tid];
                } else if (r < KDEG) {
                    int nb = idx[r];
                    v = (nb > 0) ? feat[(size_t)nb * EDIM + (tid - EDIM)] : 0.f;
                } else {
                    v = 0.f;
                }
                sS[r * H1DIM + tid] = f2bf(v);
            }
            for (int r = KDEG + 1; r < 48; ++r) sS[r * H1DIM + tid] = f2bf(0.f);
            __syncthreads();

            // ---- GEMM1: H1 = relu(S @ W1 + b1), 48x128; 6 tiles per wave ----
            float h1reg[6][8];
            for (int tt = 0; tt < 6; ++tt) {
                const int mt = tt % 3;
                const int nt = w * 2 + tt / 3;
                v8f acc = {};
                #pragma unroll
                for (int ks = 0; ks < 4; ++ks) {
                    v16bf a = ldsA(sS,  H1DIM, mt * 16, ks * 32, l);
                    v16bf b = ldsB(sW1, H1DIM, ks * 32, nt * 16, l);
                    acc = bf16_wmma(a, b, acc);
                }
                const float bv = b1[nt * 16 + (l & 15)];
                #pragma unroll
                for (int r = 0; r < 8; ++r) {
                    float v = acc[r] + bv;
                    h1reg[tt][r] = v > 0.f ? v : 0.f;
                }
            }
            __syncthreads();
            for (int tt = 0; tt < 6; ++tt) {
                const int mbase = (tt % 3) * 16 + ((l >= 16) ? 8 : 0);
                const int ncol  = (w * 2 + tt / 3) * 16 + (l & 15);
                #pragma unroll
                for (int r = 0; r < 8; ++r)
                    sS[(mbase + r) * H1DIM + ncol] = f2bf(h1reg[tt][r]);
            }
            __syncthreads();

            // ---- GEMM2: H2 = relu(H1 @ W2 + b2), 48x64; wave w owns N-tile w ----
            {
                const int n0 = w * 16;
                for (int mt = 0; mt < 3; ++mt) {
                    v8f acc = {};
                    #pragma unroll
                    for (int ks = 0; ks < 4; ++ks) {
                        v16bf a = ldsA(sS, H1DIM, mt * 16, ks * 32, l);
                        v16bf b;
                        const int kb = ks * 32 + ((l >= 16) ? 16 : 0);
                        const int n  = n0 + (l & 15);
                        #pragma unroll
                        for (int j = 0; j < 16; ++j)
                            b[j] = f2bf(W2[(kb + j) * H2DIM + n]);   // L2-resident
                        acc = bf16_wmma(a, b, acc);
                    }
                    const float bv = b2[n0 + (l & 15)];
                    const int mbase = mt * 16 + ((l >= 16) ? 8 : 0);
                    #pragma unroll
                    for (int r = 0; r < 8; ++r) {
                        float v = acc[r] + bv;
                        sH2[(mbase + r) * H2DIM + n0 + (l & 15)] = v > 0.f ? v : 0.f;
                    }
                }
            }
            __syncthreads();

            // ---- lk head ----
            if (tid < KDEG + 1) {
                float acc = blk[0];
                for (int h = 0; h < H2DIM; ++h) acc += sH2[tid * H2DIM + h] * Wlk[h];
                sLk[tid] = acc;
            }
            __syncthreads();

            // ---- control: masked argmax, action head, pop ----
            if (tid == 0) {
                float best = -3.4e38f; int ut = 0;
                for (int r = 0; r <= KDEG; ++r) {
                    float v = (r < KDEG && !sValid[r]) ? -3.4e38f : sLk[r];
                    if (v > best) { best = v; ut = r; }
                }
                const int nd  = (sDone || ut == KDEG) ? 1 : 0;
                const int utc = ut < KDEG ? ut : KDEG - 1;
                float a0 = bact[0], a1 = bact[1];
                for (int h = 0; h < H2DIM; ++h) {
                    float hv = sH2[utc * H2DIM + h];
                    a0 += hv * Wact[h * 2 + 0];
                    a1 += hv * Wact[h * 2 + 1];
                }
                const float take = (a1 > a0) ? 1.f : 0.f;
                sU = idx[utc];
                sTake = take;
                sUpd = nd ? 0 : 1;
                sInvDen = 1.f / (sCnt + take + 1.f);   // old cnt
                if (!nd) { sValid[utc] = 0; sCnt += take; }
                sDone = nd;
            }
            __syncthreads();
            const int   u    = sU;
            const float take = sTake;
            const float invd = sInvDen;
            const int   upd  = sUpd;

            // ---- stage hv_mean & u_orig as bf16 in LDS (coalesced, once) ----
            {
                const float* uorig = X + (size_t)u * FDIM;
                const float* iorig = X + (size_t)i * FDIM;
                for (int f = tid; f < FDIM; f += 128) {
                    float uo = uorig[f];
                    sUo[f] = f2bf(uo);
                    sHv[f] = f2bf((sumhv[f] + take * uo + iorig[f]) * invd);
                }
            }
            __syncthreads();

            // ---- embed pass: [hv_mean ; u_orig] @ W_emb as one 16x8192x64 WMMA run ----
            {
                const int n0 = w * 16;
                v8f acc = {};
                const int m = l & 15;
                for (int ks = 0; ks < FDIM / 32; ++ks) {
                    const int k0 = ks * 32;
                    v16bf a = {};
                    if (m < 2) {
                        const int kb = k0 + ((l >= 16) ? 16 : 0);
                        #pragma unroll
                        for (int j = 0; j < 16; ++j) {
                            int k = kb + j;
                            a[j] = (m == 0) ? sHv[k] : sUo[k];
                        }
                    }
                    v16bf b;
                    const int kbB = k0 + ((l >= 16) ? 16 : 0);
                    const int n   = n0 + (l & 15);
                    #pragma unroll
                    for (int j = 0; j < 16; ++j)
                        b[j] = f2bf(Wemb[(size_t)(kbB + j) * EDIM + n]);  // L2-resident
                    acc = bf16_wmma(a, b, acc);
                }
                if (l < 16 && upd) {
                    const int col = n0 + l;
                    const float bv = bemb[col];
                    float fi = acc[0] + bv; fi = fi > 0.f ? fi : 0.f;
                    float fu = acc[1] + bv; fu = fu > 0.f ? fu : 0.f;
                    feat[(size_t)i * EDIM + col] = fi;   // .at[i].set first,
                    feat[(size_t)u * EDIM + col] = fu;   // .at[u].set wins if u==i
                }
            }
            __threadfence();
            __syncthreads();

            // ---- sum_hv += take * u_orig (after embed consumed old sum) ----
            if (upd && take > 0.5f) {
                const float* uorig = X + (size_t)u * FDIM;
                for (int f = tid; f < FDIM; f += 128) sumhv[f] += uorig[f];
            }
            __syncthreads();
        }
    }
}

extern "C" void kernel_launch(void* const* d_in, const int* in_sizes, int n_in,
                              void* d_out, int out_size, void* d_ws, size_t ws_size,
                              hipStream_t stream) {
    const float* X    = (const float*)d_in[0];
    const int*   nbr  = (const int*)  d_in[1];
    const float* Wemb = (const float*)d_in[2];
    const float* bemb = (const float*)d_in[3];
    const float* W1   = (const float*)d_in[4];
    const float* b1   = (const float*)d_in[5];
    const float* W2   = (const float*)d_in[6];
    const float* b2   = (const float*)d_in[7];
    const float* Wlk  = (const float*)d_in[8];
    const float* blk  = (const float*)d_in[9];
    const float* Wact = (const float*)d_in[10];
    const float* bact = (const float*)d_in[11];

    float* feat  = (float*)d_out;   // [N,E]
    float* sumhv = (float*)d_ws;    // first 8192 floats of scratch

    // Phase 1: HBM-bound bf16 WMMA GEMM with TDM-streamed A tiles.
    embed_gemm<<<dim3((N_NODES / 16) * (EDIM / 16)), dim3(32), 0, stream>>>(X, Wemb, bemb, feat);

    // Phase 2: inherently-sequential node scan — one persistent workgroup, WMMA MLP.
    policy_scan<<<dim3(1), dim3(128), 0, stream>>>(X, nbr, Wemb, bemb, W1, b1,
                                                   W2, b2, Wlk, blk, Wact, bact,
                                                   feat, sumhv);
}
